// LavaNetwork_8538394984827
// MI455X (gfx1250) — matrix operand order repaired and checked by reading, not compiled, not run on hardware
//
#include <hip/hip_runtime.h>

// ---------------------------------------------------------------------------
// CUBA-LIF two-layer SNN for MI455X (gfx1250, wave32, WMMA).
// GEMMs use v_wmma_f32_16x16x32_bf16; all operands pre-packed into the
// documented per-lane fragment layouts so the GEMM inner loop is pure
// b128 loads (L2-resident) + WMMA, double-buffered one K-step ahead so
// loads overlap WMMAs (partial s_wait_loadcnt instead of 0).
// Memory-bound workload (~1.2GB HBM traffic vs ~103 GFLOP).
// ---------------------------------------------------------------------------

typedef __bf16 v16bf __attribute__((ext_vector_type(16)));
typedef float  v8f   __attribute__((ext_vector_type(8)));

#define N_IN   512
#define N_HID  2048
#define N_OUT  256
#define BATCH  32
#define TSTEPS 1024

static __device__ __forceinline__ unsigned short f32_to_bf16_rne(float f) {
    unsigned u = __float_as_uint(f);
    u += 0x7FFFu + ((u >> 16) & 1u);
    return (unsigned short)(u >> 16);
}

// A-matrix (16x32 bf16) per-lane K index for half h (0..15), lane-half hi.
// Lanes 0-15: V0..3 -> K 0..7, V4..7 -> K 16..23 ; lanes 16-31: K 8..15, 24..31.
static __device__ __forceinline__ unsigned a_frag_K(unsigned h, unsigned hi) {
    return h + (h < 8u ? 0u : 8u) + (hi ? 8u : 0u);
}

// ---------------------------------------------------------------------------
// Pack x (B, N_IN, T) f32 {0,1} -> A-fragment-major bf16: frag = [b][tt][kt],
// 32 lanes x 16 halves (32B) contiguous per lane. Writes fully coalesced.
// ---------------------------------------------------------------------------
__global__ void __launch_bounds__(256) pack_x_kernel(const float* __restrict__ x,
                                                     unsigned short* __restrict__ af) {
    unsigned gid  = blockIdx.x * 256u + threadIdx.x;
    unsigned lane = gid & 31u;
    unsigned frag = gid >> 5;
    unsigned kt   = frag & 15u;          // N_IN/32 = 16
    unsigned tmp  = frag >> 4;
    unsigned tt   = tmp & 63u;           // T/16 = 64
    unsigned b    = tmp >> 6;
    unsigned m    = lane & 15u, hi = lane >> 4;
    unsigned t    = tt * 16u + m;
    unsigned uo[8];
#pragma unroll
    for (int j = 0; j < 8; ++j) {
        unsigned K0 = a_frag_K(2u * j, hi);
        unsigned K1 = a_frag_K(2u * j + 1u, hi);
        float v0 = x[((size_t)b * N_IN + (kt * 32u + K0)) * TSTEPS + t];
        float v1 = x[((size_t)b * N_IN + (kt * 32u + K1)) * TSTEPS + t];
        unsigned p0 = v0 > 0.5f ? 0x3F80u : 0u;   // 1.0 bf16, exact
        unsigned p1 = v1 > 0.5f ? 0x3F80u : 0u;
        uo[j] = p0 | (p1 << 16);
    }
    uint4* dst = (uint4*)(af + (size_t)frag * 512u + lane * 16u);
    dst[0] = make_uint4(uo[0], uo[1], uo[2], uo[3]);
    dst[1] = make_uint4(uo[4], uo[5], uo[6], uo[7]);
}

// ---------------------------------------------------------------------------
// Pack W (O, I) f32 -> B-fragment-major bf16 (RNE): frag = [kt][nt].
// B (32x16 bf16): lane l holds N=l%16; K = a_frag_K(half_index, l>=16) (same order as A).
// ---------------------------------------------------------------------------
__global__ void __launch_bounds__(256) pack_w_kernel(const float* __restrict__ w,
                                                     unsigned short* __restrict__ bf,
                                                     int I, int numNT) {
    unsigned gid  = blockIdx.x * 256u + threadIdx.x;
    unsigned lane = gid & 31u;
    unsigned frag = gid >> 5;
    unsigned nt   = frag % (unsigned)numNT;
    unsigned kt   = frag / (unsigned)numNT;
    unsigned n    = lane & 15u, hi = lane >> 4;
    unsigned uo[8];
#pragma unroll
    for (int j = 0; j < 8; ++j) {
        unsigned K0 = a_frag_K(2u * j, hi);
        unsigned K1 = K0 + 1u;
        unsigned short p0 = f32_to_bf16_rne(w[(size_t)(nt * 16u + n) * I + kt * 32u + K0]);
        unsigned short p1 = f32_to_bf16_rne(w[(size_t)(nt * 16u + n) * I + kt * 32u + K1]);
        uo[j] = (unsigned)p0 | ((unsigned)p1 << 16);
    }
    uint4* dst = (uint4*)(bf + (size_t)frag * 512u + lane * 16u);
    dst[0] = make_uint4(uo[0], uo[1], uo[2], uo[3]);
    dst[1] = make_uint4(uo[4], uo[5], uo[6], uo[7]);
}

// ---------------------------------------------------------------------------
// GEMM: C[b][t][o] f32 = sum_k A[b][t][k] * W[o][k], fragments pre-packed.
// One wave owns a 32(t) x 64(o) tile: 2 A-frags x 4 B-frags per K-step,
// 8 f32 accumulators (v8f). Fragment loads double-buffered one K-step
// ahead so the compiler can overlap them with the 8 WMMAs of the current
// step. All re-reads of W / A come from the 192MB L2.
// ---------------------------------------------------------------------------
__global__ void __launch_bounds__(256) gemm_kernel(const unsigned short* __restrict__ A,
                                                   const unsigned short* __restrict__ Bm,
                                                   float* __restrict__ C,
                                                   int numMTB, int numNB, int numK, int NO) {
    unsigned wave = blockIdx.x * 8u + (threadIdx.x >> 5);
    unsigned lane = threadIdx.x & 31u;
    unsigned nb   = wave % (unsigned)numNB;
    unsigned tmp  = wave / (unsigned)numNB;
    unsigned mtb  = tmp % (unsigned)numMTB;
    unsigned b    = tmp / (unsigned)numMTB;
    unsigned numMT = (unsigned)numMTB * 2u;
    int numNT = NO / 16;

    size_t aIdx0 = ((size_t)(b * numMT + mtb * 2u)) * numK;   // fragment indices
    size_t aIdx1 = aIdx0 + numK;
    size_t laneOff = (size_t)lane * 16u;

    const unsigned short* Ap0 = A + aIdx0 * 512u + laneOff;
    const unsigned short* Ap1 = A + aIdx1 * 512u + laneOff;
    const unsigned short* Bp  = Bm + (size_t)nb * 4u * 512u + laneOff;
    size_t bStride = (size_t)numNT * 512u;   // halves per K-step of B fragments

    v8f   acc[2][4] = {};
    v16bf a0[2], a1[2], bb[2][4];

    // prologue: load K-step 0
    a0[0] = *(const v16bf*)(Ap0);
    a1[0] = *(const v16bf*)(Ap1);
#pragma unroll
    for (int j = 0; j < 4; ++j)
        bb[0][j] = *(const v16bf*)(Bp + (size_t)j * 512u);

#pragma unroll 2
    for (int kt = 0; kt < numK; ++kt) {
        int cs = kt & 1, ns = cs ^ 1;
        if (kt + 1 < numK) {   // issue next K-step's 12 b128 loads before WMMAs
            a0[ns] = *(const v16bf*)(Ap0 + (size_t)(kt + 1) * 512u);
            a1[ns] = *(const v16bf*)(Ap1 + (size_t)(kt + 1) * 512u);
#pragma unroll
            for (int j = 0; j < 4; ++j)
                bb[ns][j] = *(const v16bf*)(Bp + (size_t)(kt + 1) * bStride + (size_t)j * 512u);
        }
        if (kt + 2 < numK) {   // -> global_prefetch_b8: pull K-step+2 toward L0/L2
            __builtin_prefetch(Ap0 + (size_t)(kt + 2) * 512u, 0, 1);
            __builtin_prefetch(Ap1 + (size_t)(kt + 2) * 512u, 0, 1);
        }
#pragma unroll
        for (int j = 0; j < 4; ++j) {
            acc[0][j] = __builtin_amdgcn_wmma_f32_16x16x32_bf16(
                false, a0[cs], false, bb[cs][j], (short)0, acc[0][j], false, false);
            acc[1][j] = __builtin_amdgcn_wmma_f32_16x16x32_bf16(
                false, a1[cs], false, bb[cs][j], (short)0, acc[1][j], false, false);
        }
    }

    // C layout: lane l, VGPR r -> (M = r + 8*(l>=16), N = l%16)
    unsigned n = lane & 15u, hi = lane >> 4;
#pragma unroll
    for (int mi = 0; mi < 2; ++mi) {
        size_t rowBase = (size_t)b * TSTEPS + mtb * 32u + mi * 16u + hi * 8u;
        unsigned obase = nb * 64u + n;
#pragma unroll
        for (int j = 0; j < 4; ++j) {
#pragma unroll
            for (int r = 0; r < 8; ++r)
                C[(rowBase + r) * NO + obase + j * 16u] = acc[mi][j][r];
        }
    }
}

// ---------------------------------------------------------------------------
// LIF scan: one lane per neuron, sequential over T. Coalesced cur reads
// (cur is t-major). Writes spikes to d_out in (b,o,t) layout and (layer 1)
// directly into GEMM2's A-fragment layout via an LDS 16x32 transpose.
// ---------------------------------------------------------------------------
template <bool WRITE_FRAG>
__global__ void __launch_bounds__(256) scan_kernel(const float* __restrict__ cur,
                                                   const float* __restrict__ bias,
                                                   float* __restrict__ spkOut,
                                                   unsigned short* __restrict__ afrag,
                                                   int NO) {
    __shared__ unsigned short lds16[8][16][32];
    unsigned slot = threadIdx.x >> 5;
    unsigned lane = threadIdx.x & 31u;
    unsigned wave = blockIdx.x * 8u + slot;
    unsigned wavesPerB = (unsigned)NO / 32u;
    unsigned b     = wave / wavesPerB;
    unsigned obase = (wave % wavesPerB) * 32u;
    unsigned o     = obase + lane;
    unsigned numK  = (unsigned)NO / 32u;
    unsigned kt    = obase >> 5;

    float bv = bias[o];
    float v  = 0.0f;
    float* outp = spkOut + ((size_t)b * NO + o) * TSTEPS;
    const float* cp = cur + (size_t)b * TSTEPS * NO + o;

    for (int tc = 0; tc < TSTEPS / 16; ++tc) {
        float sv[16];
#pragma unroll
        for (int m = 0; m < 16; ++m) {
            float c = cp[(size_t)(tc * 16 + m) * NO];
            v = v * 0.95f + c + bv;
            bool s = (v >= 1.0f);
            v = s ? 0.0f : v;           // reset-to-zero on spike
            sv[m] = s ? 1.0f : 0.0f;
            if (WRITE_FRAG)
                lds16[slot][m][lane] = s ? (unsigned short)0x3F80 : (unsigned short)0;
        }
        // d_out spk (b,o,t): per-lane contiguous 64B bursts
        float4* op = (float4*)(outp + tc * 16);
        op[0] = make_float4(sv[0], sv[1], sv[2], sv[3]);
        op[1] = make_float4(sv[4], sv[5], sv[6], sv[7]);
        op[2] = make_float4(sv[8], sv[9], sv[10], sv[11]);
        op[3] = make_float4(sv[12], sv[13], sv[14], sv[15]);

        if (WRITE_FRAG) {
            __syncthreads();
            unsigned ml = lane & 15u, hi = lane >> 4;
            unsigned uo[8];
#pragma unroll
            for (int j = 0; j < 8; ++j) {
                unsigned K0 = a_frag_K(2u * j, hi);
                unsigned K1 = a_frag_K(2u * j + 1u, hi);
                uo[j] = (unsigned)lds16[slot][ml][K0] |
                        ((unsigned)lds16[slot][ml][K1] << 16);
            }
            size_t frag = ((size_t)(b * (TSTEPS / 16) + tc)) * numK + kt;
            uint4* dst = (uint4*)(afrag + frag * 512u + lane * 16u);
            dst[0] = make_uint4(uo[0], uo[1], uo[2], uo[3]);
            dst[1] = make_uint4(uo[4], uo[5], uo[6], uo[7]);
            __syncthreads();
        }
    }
}

extern "C" void kernel_launch(void* const* d_in, const int* in_sizes, int n_in,
                              void* d_out, int out_size, void* d_ws, size_t ws_size,
                              hipStream_t stream) {
    const float* x  = (const float*)d_in[0];
    const float* w1 = (const float*)d_in[1];
    const float* b1 = (const float*)d_in[2];
    const float* w2 = (const float*)d_in[3];
    const float* b2 = (const float*)d_in[4];

    float* out_spk1 = (float*)d_out;
    float* out_spk2 = out_spk1 + (size_t)BATCH * N_HID * TSTEPS;

    char* ws = (char*)d_ws;
    unsigned short* a1f = (unsigned short*)(ws);                         // 32 MB  x fragments
    unsigned short* w1f = (unsigned short*)(ws + (33u << 20));           // 2 MB   w1 fragments
    unsigned short* w2f = (unsigned short*)(ws + (35u << 20));           // 1 MB   w2 fragments
    float*          curb = (float*)(ws + (36u << 20));                   // 256 MB cur1 (reused as cur2)
    unsigned short* a2f = (unsigned short*)(ws + (36u << 20) + ((size_t)256 << 20)); // 128 MB spk1 fragments

    // Fragment packing (one pass, fully coalesced writes)
    pack_x_kernel<<<4096, 256, 0, stream>>>(x, a1f);
    pack_w_kernel<<<256, 256, 0, stream>>>(w1, w1f, N_IN, N_HID / 16);
    pack_w_kernel<<<128, 256, 0, stream>>>(w2, w2f, N_HID, N_OUT / 16);

    // Layer 1: GEMM (M=T per batch, N=2048, K=512) -> cur1, then LIF scan
    // waves = 32 * (1024/32) * (2048/64) = 32768 -> 4096 blocks of 8 waves
    gemm_kernel<<<4096, 256, 0, stream>>>(a1f, w1f, curb, 32, 32, 16, N_HID);
    scan_kernel<true><<<256, 256, 0, stream>>>(curb, b1, out_spk1, a2f, N_HID);

    // Layer 2: GEMM (N=256, K=2048) -> cur2 (aliases cur1 buffer), then scan
    // waves = 32 * 32 * 4 = 4096 -> 512 blocks
    gemm_kernel<<<512, 256, 0, stream>>>(a2f, w2f, curb, 32, 4, 64, N_OUT);
    scan_kernel<false><<<32, 256, 0, stream>>>(curb, b2, out_spk2, nullptr, N_OUT);
}
